// GroupedQueryAttention_36962488549556
// MI455X (gfx1250) — matrix operand order repaired and checked by reading, not compiled, not run on hardware
//
#include <hip/hip_runtime.h>
#include <hip/hip_bf16.h>
#include <math.h>

typedef __bf16 bf16;
typedef __bf16 v8bf  __attribute__((ext_vector_type(8)));
typedef __bf16 v16bf __attribute__((ext_vector_type(16)));
typedef float  v8f   __attribute__((ext_vector_type(8)));

#define SEQ     2048
#define EMB     4096
#define KVD     1024
#define HEADS   32
#define KVHEADS 8
#define HD      128

// D = A(16x32 bf16) * B(32x16 bf16) + C(16x16 f32)
__device__ __forceinline__ v8f wmma_bf16f32(v16bf a, v16bf b, v8f c) {
  return __builtin_amdgcn_wmma_f32_16x16x32_bf16(false, a, false, b, (short)0, c,
                                                 false, false);
}

__device__ __forceinline__ v16bf cat8(v8bf lo, v8bf hi) {
  return __builtin_shufflevector(lo, hi, 0, 1, 2, 3, 4, 5, 6, 7,
                                         8, 9, 10, 11, 12, 13, 14, 15);
}

// 16B global -> LDS async copy (CDNA5 TDM-free async path, ASYNCcnt-tracked).
// LDS dest and global src must be 16B aligned.
__device__ __forceinline__ void copy16_g2l(bf16* lds, const bf16* g) {
#if defined(__gfx1250__)
  unsigned loff = (unsigned)(size_t)((__attribute__((address_space(3))) bf16*)lds);
  asm volatile("global_load_async_to_lds_b128 %0, %1, off"
               :
               : "v"(loff), "v"(g)
               : "memory");
#else
  *(uint4*)lds = *(const uint4*)g;
#endif
}

// Wait for this wave's outstanding async LDS copies before the barrier.
__device__ __forceinline__ void async_fence() {
#if defined(__gfx1250__)
  asm volatile("s_wait_asynccnt 0" ::: "memory");
#endif
}

// ---------------------------------------------------------------- fp32 -> bf16
__global__ void k_cvt(const float* __restrict__ in, bf16* __restrict__ out, int n) {
  int stride = blockDim.x * gridDim.x;
  for (int i = blockIdx.x * blockDim.x + threadIdx.x; i < n; i += stride)
    out[i] = (bf16)in[i];
}

// -------------------------------------------------- V: [s][kv*128+d] -> [kv*128+d][s]
__global__ void k_transpose_v(const bf16* __restrict__ Vb, bf16* __restrict__ Vt) {
  int idx = blockIdx.x * blockDim.x + threadIdx.x;  // < SEQ*KVD
  int s = idx >> 10;
  int c = idx & (KVD - 1);
  Vt[(size_t)c * SEQ + s] = Vb[idx];
}

// ---------------------------------------------------------------- NT GEMM
// C[M,N] = A[M,K] * B[N,K]^T + bias[N].  A,B bf16 row-major (K contiguous).
// Block: 128x128 tile, 256 threads (8 waves, 2x4), K-step 32 through LDS.
template <bool OUT_F32>
__global__ __launch_bounds__(256)
void k_gemm_nt(const bf16* __restrict__ A, const bf16* __restrict__ B,
               const float* __restrict__ bias, void* __restrict__ Cout,
               int M, int N, int K) {
  __shared__ bf16 As[128][40];  // pad 8 bf16 (16B) -> 80B rows, 16B aligned
  __shared__ bf16 Bs[128][40];

  const int m0 = blockIdx.y * 128;
  const int n0 = blockIdx.x * 128;
  const int tid  = threadIdx.x;
  const int w    = tid >> 5;
  const int lane = tid & 31;
  const int lr   = lane & 15;   // row/col within 16-lane half
  const int hs   = lane >> 4;   // which half of the wave
  const int wm   = (w >> 2) * 64;
  const int wn   = (w & 3) * 32;
  const int ldr  = tid >> 1;          // 0..127: tile row loaded by this thread
  const int ldc  = (tid & 1) * 16;    // 0 or 16: K-half loaded by this thread

  v8f acc[4][2] = {};

  for (int kk = 0; kk < K; kk += 32) {
    __syncthreads();
    {
      const bf16* ag = A + (size_t)(m0 + ldr) * K + kk + ldc;
      copy16_g2l(&As[ldr][ldc],     ag);
      copy16_g2l(&As[ldr][ldc + 8], ag + 8);
      const bf16* bg = B + (size_t)(n0 + ldr) * K + kk + ldc;
      copy16_g2l(&Bs[ldr][ldc],     bg);
      copy16_g2l(&Bs[ldr][ldc + 8], bg + 8);
    }
    async_fence();
    __syncthreads();

    v16bf bfrag[2];
#pragma unroll
    for (int nt = 0; nt < 2; ++nt) {
      const bf16* p = &Bs[wn + nt * 16 + lr][hs * 8];
      bfrag[nt] = cat8(*(const v8bf*)p, *(const v8bf*)(p + 16));
    }
#pragma unroll
    for (int mt = 0; mt < 4; ++mt) {
      const bf16* p = &As[wm + mt * 16 + lr][hs * 8];
      v16bf af = cat8(*(const v8bf*)p, *(const v8bf*)(p + 16));
#pragma unroll
      for (int nt = 0; nt < 2; ++nt)
        acc[mt][nt] = wmma_bf16f32(af, bfrag[nt], acc[mt][nt]);
    }
  }

#pragma unroll
  for (int mt = 0; mt < 4; ++mt) {
#pragma unroll
    for (int nt = 0; nt < 2; ++nt) {
      const int col   = n0 + wn + nt * 16 + lr;
      const float bv  = bias[col];
      const int rbase = m0 + wm + mt * 16 + hs * 8;
#pragma unroll
      for (int j = 0; j < 8; ++j) {
        float v = acc[mt][nt][j] + bv;
        size_t off = (size_t)(rbase + j) * N + col;
        if (OUT_F32) ((float*)Cout)[off] = v;
        else         ((bf16*)Cout)[off]  = (bf16)v;
      }
    }
  }
}

// ---------------------------------------------------------------- attention
// Flash-style online base-2 softmax (softermax). Block = 1 head x 64 q-rows,
// 4 waves x 16 rows each, streaming 64-key blocks through LDS.
__global__ __launch_bounds__(128)
void k_attention(const bf16* __restrict__ Qb, const bf16* __restrict__ Kb,
                 const bf16* __restrict__ Vt, bf16* __restrict__ Cb) {
  __shared__ bf16 Ks[64][136];   // [key][d], pad -> 272B rows
  __shared__ bf16 Vs[128][72];   // [d][key], pad -> 144B rows
  __shared__ bf16 Ps[4][16][72]; // per-wave P tile [qrow][key]

  const int qb  = blockIdx.x * 64;
  const int h   = blockIdx.y;
  const int kvh = h >> 2;
  const int tid = threadIdx.x;
  const int w    = tid >> 5;
  const int lane = tid & 31;
  const int lr   = lane & 15;
  const int hs   = lane >> 4;
  const int q0   = qb + w * 16;

  // resident Q fragments: 16 rows x 128 d, 4 A-frags over K=32 steps
  v16bf qf[4];
  {
    const bf16* qrow = Qb + (size_t)(q0 + lr) * EMB + h * HD;
#pragma unroll
    for (int kk = 0; kk < 4; ++kk) {
      const bf16* p = qrow + kk * 32 + hs * 8;
      qf[kk] = cat8(*(const v8bf*)p, *(const v8bf*)(p + 16));
    }
  }

  v8f ctx[8] = {};
  v8f mrow, lrow;
#pragma unroll
  for (int j = 0; j < 8; ++j) { mrow[j] = -1e30f; lrow[j] = 0.0f; }

  const float scale = 0.088388347648318447f;  // 1/sqrt(128)

  for (int kb = 0; kb < SEQ; kb += 64) {
    __syncthreads();
    {
      const int r  = tid >> 1;
      const int hh = (tid & 1) * 64;
      const bf16* ksrc = Kb + (size_t)(kb + r) * KVD + kvh * HD + hh;
#pragma unroll
      for (int u = 0; u < 8; ++u)
        copy16_g2l(&Ks[r][hh + u * 8], ksrc + u * 8);
      const bf16* vsrc = Vt + (size_t)(kvh * HD + tid) * SEQ + kb;
#pragma unroll
      for (int u = 0; u < 8; ++u)
        copy16_g2l(&Vs[tid][u * 8], vsrc + u * 8);
    }
    async_fence();
    __syncthreads();

    // S = scale * Q K^T  -> 16 x 64 per wave (4 N-tiles)
    v8f sfr[4] = {};
#pragma unroll
    for (int nt = 0; nt < 4; ++nt) {
#pragma unroll
      for (int kk = 0; kk < 4; ++kk) {
        const bf16* p = &Ks[nt * 16 + lr][kk * 32 + hs * 8];
        v16bf kf = cat8(*(const v8bf*)p, *(const v8bf*)(p + 16));
        sfr[nt] = wmma_bf16f32(qf[kk], kf, sfr[nt]);
      }
#pragma unroll
      for (int j = 0; j < 8; ++j) sfr[nt][j] *= scale;
    }

    // block row-max (reduce across the 16-lane half holding each row)
    v8f mloc = sfr[0];
#pragma unroll
    for (int nt = 1; nt < 4; ++nt)
#pragma unroll
      for (int j = 0; j < 8; ++j) mloc[j] = fmaxf(mloc[j], sfr[nt][j]);
#pragma unroll
    for (int d = 8; d >= 1; d >>= 1)
#pragma unroll
      for (int j = 0; j < 8; ++j)
        mloc[j] = fmaxf(mloc[j], __shfl_xor(mloc[j], d, 32));

    v8f mnew, corr;
#pragma unroll
    for (int j = 0; j < 8; ++j) {
      mnew[j] = fmaxf(mrow[j], mloc[j]);
      corr[j] = exp2f(mrow[j] - mnew[j]);
    }

    // P = exp2(S - mnew); stash as bf16 in LDS in A-operand order; row sums
    v8f psum = {};
#pragma unroll
    for (int nt = 0; nt < 4; ++nt)
#pragma unroll
      for (int j = 0; j < 8; ++j) {
        float pv = exp2f(sfr[nt][j] - mnew[j]);
        psum[j] += pv;
        Ps[w][j + hs * 8][nt * 16 + lr] = (bf16)pv;
      }
#pragma unroll
    for (int d = 8; d >= 1; d >>= 1)
#pragma unroll
      for (int j = 0; j < 8; ++j)
        psum[j] += __shfl_xor(psum[j], d, 32);

#pragma unroll
    for (int j = 0; j < 8; ++j) {
      lrow[j] = lrow[j] * corr[j] + psum[j];
      mrow[j] = mnew[j];
    }
#pragma unroll
    for (int dt = 0; dt < 8; ++dt)
#pragma unroll
      for (int j = 0; j < 8; ++j) ctx[dt][j] *= corr[j];

    // ctx += P * V  (K-dim = 64 keys, 2 steps of 32)
#pragma unroll
    for (int kk = 0; kk < 2; ++kk) {
      const bf16* pp = &Ps[w][lr][kk * 32 + hs * 8];
      v16bf pf = cat8(*(const v8bf*)pp, *(const v8bf*)(pp + 16));
#pragma unroll
      for (int dt = 0; dt < 8; ++dt) {
        const bf16* vp = &Vs[dt * 16 + lr][kk * 32 + hs * 8];
        v16bf vf = cat8(*(const v8bf*)vp, *(const v8bf*)(vp + 16));
        ctx[dt] = wmma_bf16f32(pf, vf, ctx[dt]);
      }
    }
  }

  // normalize and write context in [s][h*128+d] layout (bf16)
#pragma unroll
  for (int dt = 0; dt < 8; ++dt) {
#pragma unroll
    for (int j = 0; j < 8; ++j) {
      float o = ctx[dt][j] / lrow[j];
      Cb[(size_t)(q0 + j + hs * 8) * EMB + h * HD + dt * 16 + lr] = (bf16)o;
    }
  }
}

// ---------------------------------------------------------------- launch
extern "C" void kernel_launch(void* const* d_in, const int* in_sizes, int n_in,
                              void* d_out, int out_size, void* d_ws, size_t ws_size,
                              hipStream_t stream) {
  const float* x     = (const float*)d_in[0];
  const float* qw    = (const float*)d_in[1];
  const float* qbias = (const float*)d_in[2];
  const float* kw    = (const float*)d_in[3];
  const float* kbias = (const float*)d_in[4];
  const float* vw    = (const float*)d_in[5];
  const float* vbias = (const float*)d_in[6];
  const float* ow    = (const float*)d_in[7];
  const float* obias = (const float*)d_in[8];

  char* ws = (char*)d_ws;
  size_t off = 0;
  auto alloc = [&](size_t nelem) { bf16* p = (bf16*)(ws + off); off += nelem * 2; return p; };
  bf16* xb  = alloc((size_t)SEQ * EMB);
  bf16* qwb = alloc((size_t)EMB * EMB);
  bf16* kwb = alloc((size_t)KVD * EMB);
  bf16* vwb = alloc((size_t)KVD * EMB);
  bf16* owb = alloc((size_t)EMB * EMB);
  bf16* Qb  = alloc((size_t)SEQ * EMB);
  bf16* Kb  = alloc((size_t)SEQ * KVD);
  bf16* Vb  = alloc((size_t)SEQ * KVD);
  bf16* Vt  = alloc((size_t)SEQ * KVD);
  bf16* Cb  = alloc((size_t)SEQ * EMB);

  k_cvt<<<4096, 256, 0, stream>>>(x,  xb,  SEQ * EMB);
  k_cvt<<<4096, 256, 0, stream>>>(qw, qwb, EMB * EMB);
  k_cvt<<<4096, 256, 0, stream>>>(kw, kwb, KVD * EMB);
  k_cvt<<<4096, 256, 0, stream>>>(vw, vwb, KVD * EMB);
  k_cvt<<<4096, 256, 0, stream>>>(ow, owb, EMB * EMB);

  // Q/K/V projections (bf16 out, bias fused)
  k_gemm_nt<false><<<dim3(EMB / 128, SEQ / 128), 256, 0, stream>>>(
      xb, qwb, qbias, Qb, SEQ, EMB, EMB);
  k_gemm_nt<false><<<dim3(KVD / 128, SEQ / 128), 256, 0, stream>>>(
      xb, kwb, kbias, Kb, SEQ, KVD, EMB);
  k_gemm_nt<false><<<dim3(KVD / 128, SEQ / 128), 256, 0, stream>>>(
      xb, vwb, vbias, Vb, SEQ, KVD, EMB);

  k_transpose_v<<<(SEQ * KVD) / 256, 256, 0, stream>>>(Vb, Vt);

  k_attention<<<dim3(SEQ / 64, HEADS), 128, 0, stream>>>(Qb, Kb, Vt, Cb);

  // O projection (f32 out, bias fused) straight into d_out
  k_gemm_nt<true><<<dim3(EMB / 128, SEQ / 128), 256, 0, stream>>>(
      Cb, owb, obias, d_out, SEQ, EMB, EMB);

  (void)in_sizes; (void)n_in; (void)out_size; (void)ws_size;
}
